// TransformerBlock_22402549416170
// MI455X (gfx1250) — matrix operand order repaired and checked by reading, not compiled, not run on hardware
//
#include <hip/hip_runtime.h>

#define BSZ 2
#define SEQ 2048
#define DIM 1024
#define NH  16
#define DH  64
#define FF  4096

typedef __bf16 v8bf  __attribute__((ext_vector_type(8)));
typedef __bf16 v16bf __attribute__((ext_vector_type(16)));
typedef float  v8f   __attribute__((ext_vector_type(8)));
typedef unsigned int v4u __attribute__((ext_vector_type(4)));
typedef int   v4i __attribute__((ext_vector_type(4)));
typedef int   v8i __attribute__((ext_vector_type(8)));

union bf16x16 { v16bf v; v8bf h[2]; };

// --- feature detection for the Tensor Data Mover path ----------------------
#if defined(__has_builtin)
# if __has_builtin(__builtin_amdgcn_tensor_load_to_lds)
#  define USE_TDM 1
# endif
#endif
#ifndef USE_TDM
# define USE_TDM 0
#endif
#if __has_include(<hip/amd_detail/amd_gfx1250_TDM.h>)
# define TDM_SIX_ARGS 1
#else
# define TDM_SIX_ARGS 0
#endif

static __device__ __forceinline__ void wait_tensorcnt0() {
#if defined(__has_builtin) && __has_builtin(__builtin_amdgcn_s_wait_tensorcnt)
    __builtin_amdgcn_s_wait_tensorcnt(0);
#else
    __asm__ volatile("s_wait_tensorcnt 0" ::: "memory");
#endif
}

static __device__ __forceinline__ __bf16 f2bf(float f) {
    unsigned int u = __builtin_bit_cast(unsigned int, f);
    unsigned int r = u + 0x7FFFu + ((u >> 16) & 1u);
    unsigned short s = (unsigned short)(r >> 16);
    return __builtin_bit_cast(__bf16, s);
}

// ---------------------------------------------------------------------------
// Generic 2D TDM tile load: tile_w contiguous elements per row, tile_h rows,
// row stride `stride` (elements, bf16), optional LDS pad every (2<<pad_i)
// DWORDs of (pad_a+1) DWORDs -> bank-conflict-free fragment reads.
// ---------------------------------------------------------------------------
#if USE_TDM
static __device__ __forceinline__ void tdm_load_2d(const __bf16* base,
        int tile_w, int tile_h, int stride, int tw, int th,
        int pad_i, int pad_a, unsigned lds_off) {
    unsigned long long ga = (unsigned long long)(uintptr_t)base;
    v4u g0 = { 1u, lds_off, (unsigned)ga,
               ((unsigned)(ga >> 32) & 0x01FFFFFFu) | 0x80000000u };
    v8i g1;
    g1[0] = (1 << 16) | (1 << 20) | (pad_i << 22) | (pad_a << 25); // 2B, pad_en
    g1[1] = (tw & 0xFFFF) << 16;                                   // tensor_dim0 lo
    g1[2] = ((unsigned)tw >> 16) | ((th & 0xFFFF) << 16);          // |tensor_dim1 lo
    g1[3] = ((unsigned)th >> 16) | (tile_w << 16);                 // |tile_dim0
    g1[4] = tile_h;                                                // tile_dim1, dim2=0
    g1[5] = stride;                                                // dim0_stride lo32
    g1[6] = 0;
    g1[7] = 0;
    v4i z4 = {};
#if TDM_SIX_ARGS
    v8i z8 = {};
    __builtin_amdgcn_tensor_load_to_lds(g0, g1, z4, z4, z8, 0);
#else
    __builtin_amdgcn_tensor_load_to_lds(g0, g1, z4, z4, 0);
#endif
}
#endif

// ---------------------------------------------------------------------------
// Elementwise helpers
// ---------------------------------------------------------------------------
__global__ void cast_f32_bf16(const float* __restrict__ src, __bf16* __restrict__ dst, int n) {
    int i = blockIdx.x * 256 + threadIdx.x;
    if (i < n) dst[i] = f2bf(src[i]);
}

__global__ void transpose_cast(const float* __restrict__ W, __bf16* __restrict__ Wt, int R, int C) {
    int idx = blockIdx.x * 256 + threadIdx.x;
    if (idx < R * C) {
        int r = idx / C, c = idx % C;
        Wt[(size_t)c * R + r] = f2bf(W[idx]);
    }
}

__global__ void transpose_v(const __bf16* __restrict__ v, __bf16* __restrict__ vt) {
    int idx = blockIdx.x * 256 + threadIdx.x;
    int d0 = idx % DIM;
    int s  = (idx / DIM) % SEQ;
    int b  = idx / (DIM * SEQ);
    int h  = d0 / DH, d = d0 % DH;
    vt[((size_t)(b * NH + h) * DH + d) * SEQ + s] = v[idx];
}

__global__ void add_ln_kernel(const float* __restrict__ a, const float* __restrict__ b,
                              const float* __restrict__ g, const float* __restrict__ beta,
                              float* __restrict__ yf, __bf16* __restrict__ yb) {
    __shared__ float red[256];
    int row = blockIdx.x;
    const float* ar = a + (size_t)row * DIM;
    const float* br = b + (size_t)row * DIM;
    float v[4];
    float s = 0.f;
    #pragma unroll
    for (int i = 0; i < 4; ++i) {
        int c = threadIdx.x + i * 256;
        v[i] = ar[c] + br[c];
        s += v[i];
    }
    red[threadIdx.x] = s; __syncthreads();
    for (int st = 128; st > 0; st >>= 1) {
        if (threadIdx.x < st) red[threadIdx.x] += red[threadIdx.x + st];
        __syncthreads();
    }
    float mu = red[0] * (1.f / DIM);
    __syncthreads();
    float s2 = 0.f;
    #pragma unroll
    for (int i = 0; i < 4; ++i) { float d = v[i] - mu; s2 += d * d; }
    red[threadIdx.x] = s2; __syncthreads();
    for (int st = 128; st > 0; st >>= 1) {
        if (threadIdx.x < st) red[threadIdx.x] += red[threadIdx.x + st];
        __syncthreads();
    }
    float rstd = rsqrtf(red[0] * (1.f / DIM) + 1e-12f);
    #pragma unroll
    for (int i = 0; i < 4; ++i) {
        int c = threadIdx.x + i * 256;
        float y = (v[i] - mu) * rstd * g[c] + beta[c];
        size_t off = (size_t)row * DIM + c;
        yf[off] = y;
        if (yb) yb[off] = f2bf(y);
    }
}

// ---------------------------------------------------------------------------
// GEMM: C[MxN] = A[MxK](bf16 row-major) * Bt[NxK](bf16, B transposed)
// block = 256 threads = 8 waves; wave computes 16x64; block computes 128x64.
// B tile (64x32) double-buffered in LDS via TDM DMA; all 4 B fragments are
// loaded into distinct registers before the 4 WMMAs so ds_loads batch up.
// __launch_bounds__(256,1): give the scheduler VGPR headroom to keep all
// four B fragments live instead of serializing ds_load->wait->wmma.
// ---------------------------------------------------------------------------
#define LDSB 40  // padded LDS row stride in bf16 elements (32 data + 8 pad)

__global__ __launch_bounds__(256, 1)
void gemm_bf16(const __bf16* __restrict__ A, const __bf16* __restrict__ Bt,
               const float* __restrict__ bias, float* __restrict__ Cf,
               __bf16* __restrict__ Cb, int M, int N, int K,
               float scale, int gelu) {
    __shared__ __attribute__((aligned(64))) __bf16 bsm[2][64 * LDSB];
    int warp = threadIdx.x >> 5, lane = threadIdx.x & 31;
    int ln = lane & 15, half = lane >> 4;
    int m0 = blockIdx.y * 128 + warp * 16;
    int n0 = blockIdx.x * 64;
    int nk = K >> 5;
    v8f acc[4] = {};
    const __bf16* arow = A + (size_t)(m0 + ln) * K;

#if USE_TDM
    if (threadIdx.x < 32)
        tdm_load_2d(Bt + (size_t)n0 * K, 32, 64, K, K, N, 3, 3,
                    (unsigned)(uintptr_t)&bsm[0][0]);
#else
    int brow = threadIdx.x >> 2, bcol = (threadIdx.x & 3) * 8;
    {
        v8bf breg = *(const v8bf*)(Bt + (size_t)(n0 + brow) * K + bcol);
        *(v8bf*)(&bsm[0][brow * LDSB + bcol]) = breg;
    }
#endif

    for (int kt = 0; kt < nk; ++kt) {
        int k0 = kt << 5;
#if USE_TDM
        if (threadIdx.x < 32) wait_tensorcnt0();
        __syncthreads();
        if (threadIdx.x < 32 && kt + 1 < nk)
            tdm_load_2d(Bt + (size_t)n0 * K + k0 + 32, 32, 64, K, K, N, 3, 3,
                        (unsigned)(uintptr_t)&bsm[(kt + 1) & 1][0]);
#else
        __syncthreads();
        v8bf nreg;
        if (kt + 1 < nk)
            nreg = *(const v8bf*)(Bt + (size_t)(n0 + brow) * K + k0 + 32 + bcol);
#endif
        const __bf16* bb = &bsm[kt & 1][0];
        bf16x16 af;
        af.h[0] = *(const v8bf*)(arow + k0 + half * 8);
        af.h[1] = *(const v8bf*)(arow + k0 + 16 + half * 8);
        bf16x16 bfr[4];
        #pragma unroll
        for (int j = 0; j < 4; ++j) {
            const __bf16* bp = bb + (j * 16 + ln) * LDSB + half * 8;
            bfr[j].h[0] = *(const v8bf*)(bp);
            bfr[j].h[1] = *(const v8bf*)(bp + 16);
        }
        #pragma unroll
        for (int j = 0; j < 4; ++j)
            acc[j] = __builtin_amdgcn_wmma_f32_16x16x32_bf16(
                false, af.v, false, bfr[j].v, (short)0, acc[j], false, false);
#if !USE_TDM
        if (kt + 1 < nk)
            *(v8bf*)(&bsm[(kt + 1) & 1][brow * LDSB + bcol]) = nreg;
#endif
    }

    #pragma unroll
    for (int j = 0; j < 4; ++j) {
        int n = n0 + j * 16 + ln;
        float bv = bias ? bias[n] : 0.f;
        #pragma unroll
        for (int r = 0; r < 8; ++r) {
            int m = m0 + r + 8 * half;
            float y = acc[j][r] + bv;
            if (gelu) y = 0.5f * y * (1.f + erff(y * 0.70710678118654752f));
            y *= scale;
            size_t off = (size_t)m * N + n;
            if (Cf) Cf[off] = y;
            if (Cb) Cb[off] = f2bf(y);
        }
    }
}

// ---------------------------------------------------------------------------
// Fused flash-style attention.
// 4 waves/block share one (b,h) and 4 consecutive 16-query tiles, so K/V
// tiles are identical across waves: stage them in LDS via double-buffered TDM
// (two descriptors per 32-key step, overlapped with compute). Online softmax
// with half-wave shfl reductions; P bounced through per-wave LDS to become
// the ctx WMMA's A operand. q is pre-scaled by 1/sqrt(DH).
// ---------------------------------------------------------------------------
#define ATT_WARPS 4
#define KSTR 72   // 64 data + 8 pad elements (144B rows)
#define VSTR 40   // 32 data + 8 pad elements (80B rows)

__global__ __launch_bounds__(ATT_WARPS * 32, 1)
void attn_kernel(const __bf16* __restrict__ qb, const __bf16* __restrict__ kb,
                 const __bf16* __restrict__ vt, const int* __restrict__ amask,
                 __bf16* __restrict__ ctx) {
    __shared__ __attribute__((aligned(64))) __bf16 ksm[2][32 * KSTR];
    __shared__ __attribute__((aligned(64))) __bf16 vsm[2][64 * VSTR];
    __shared__ __attribute__((aligned(32))) __bf16 pbuf[ATT_WARPS][16 * 32];
    int warp = threadIdx.x >> 5, lane = threadIdx.x & 31;
    int ln = lane & 15, half = lane >> 4;
    int blk = blockIdx.x;
    int qblk = blk & 31;                    // 32 blocks per head, 4 q-tiles each
    int h = (blk >> 5) & (NH - 1);
    int b = blk >> 9;
    int q0 = (qblk * ATT_WARPS + warp) * 16;
    __bf16* pw = pbuf[warp];
    const __bf16* kbase0 = kb + (size_t)(b * SEQ) * DIM + h * DH;
    const __bf16* vbase0 = vt + (size_t)(b * NH + h) * DH * SEQ;

    bf16x16 aq[2];
    const __bf16* qrow = qb + ((size_t)(b * SEQ + q0 + ln)) * DIM + h * DH;
    #pragma unroll
    for (int c = 0; c < 2; ++c) {
        aq[c].h[0] = *(const v8bf*)(qrow + c * 32 + half * 8);
        aq[c].h[1] = *(const v8bf*)(qrow + c * 32 + 16 + half * 8);
    }

    float mrow[8], lrow[8];
    v8f acc[4] = {};
    #pragma unroll
    for (int r = 0; r < 8; ++r) { mrow[r] = -1e30f; lrow[r] = 0.f; }

#if USE_TDM
    if (threadIdx.x < 32) {
        tdm_load_2d(kbase0, 64, 32, DIM, DIM, SEQ, 4, 3, (unsigned)(uintptr_t)&ksm[0][0]);
        tdm_load_2d(vbase0, 32, 64, SEQ, SEQ, DH, 3, 3, (unsigned)(uintptr_t)&vsm[0][0]);
    }
#endif

    for (int kt = 0; kt < SEQ / 32; ++kt) {
        int key0 = kt * 32;
#if USE_TDM
        if (threadIdx.x < 32) wait_tensorcnt0();   // K/V tiles for kt landed
        __syncthreads();
        if (threadIdx.x < 32 && kt + 1 < SEQ / 32) {
            int nk0 = key0 + 32;
            tdm_load_2d(kbase0 + (size_t)nk0 * DIM, 64, 32, DIM, DIM, SEQ, 4, 3,
                        (unsigned)(uintptr_t)&ksm[(kt + 1) & 1][0]);
            tdm_load_2d(vbase0 + nk0, 32, 64, SEQ, SEQ, DH, 3, 3,
                        (unsigned)(uintptr_t)&vsm[(kt + 1) & 1][0]);
        }
#else
        __syncthreads();
        {
            int t = threadIdx.x;
            int kr = t >> 2, kc = (t & 3) * 16;
            const __bf16* ksrc = kbase0 + (size_t)(key0 + kr) * DIM + kc;
            *(v8bf*)&ksm[kt & 1][kr * KSTR + kc]     = *(const v8bf*)ksrc;
            *(v8bf*)&ksm[kt & 1][kr * KSTR + kc + 8] = *(const v8bf*)(ksrc + 8);
            int vr = t >> 1, vc = (t & 1) * 16;
            const __bf16* vsrc = vbase0 + (size_t)vr * SEQ + key0 + vc;
            *(v8bf*)&vsm[kt & 1][vr * VSTR + vc]     = *(const v8bf*)vsrc;
            *(v8bf*)&vsm[kt & 1][vr * VSTR + vc + 8] = *(const v8bf*)(vsrc + 8);
        }
        __syncthreads();
#endif
        const __bf16* kk = &ksm[kt & 1][0];
        const __bf16* vv = &vsm[kt & 1][0];

        // score tiles (16x16 each, keys key0..+15 and key0+16..+31)
        bf16x16 bk[2][2];
        #pragma unroll
        for (int t = 0; t < 2; ++t)
            #pragma unroll
            for (int c = 0; c < 2; ++c) {
                const __bf16* bp = kk + (t * 16 + ln) * KSTR + c * 32 + half * 8;
                bk[t][c].h[0] = *(const v8bf*)(bp);
                bk[t][c].h[1] = *(const v8bf*)(bp + 16);
            }
        v8f s0 = {}, s1 = {};
        s0 = __builtin_amdgcn_wmma_f32_16x16x32_bf16(false, aq[0].v, false, bk[0][0].v, (short)0, s0, false, false);
        s1 = __builtin_amdgcn_wmma_f32_16x16x32_bf16(false, aq[0].v, false, bk[1][0].v, (short)0, s1, false, false);
        s0 = __builtin_amdgcn_wmma_f32_16x16x32_bf16(false, aq[1].v, false, bk[0][1].v, (short)0, s0, false, false);
        s1 = __builtin_amdgcn_wmma_f32_16x16x32_bf16(false, aq[1].v, false, bk[1][1].v, (short)0, s1, false, false);

        int mk0 = amask[b * SEQ + key0 + ln];
        int mk1 = amask[b * SEQ + key0 + 16 + ln];
        #pragma unroll
        for (int r = 0; r < 8; ++r) {
            if (!mk0) s0[r] = -1e30f;
            if (!mk1) s1[r] = -1e30f;
        }
        float alpha[8];
        #pragma unroll
        for (int r = 0; r < 8; ++r) {
            float t = fmaxf(s0[r], s1[r]);
            #pragma unroll
            for (int msk = 8; msk >= 1; msk >>= 1) t = fmaxf(t, __shfl_xor(t, msk, 32));
            float mn = fmaxf(mrow[r], t);
            alpha[r] = __expf(mrow[r] - mn);
            float p0 = __expf(s0[r] - mn);
            float p1 = __expf(s1[r] - mn);
            s0[r] = p0; s1[r] = p1;
            float sum = p0 + p1;
            #pragma unroll
            for (int msk = 8; msk >= 1; msk >>= 1) sum += __shfl_xor(sum, msk, 32);
            lrow[r] = lrow[r] * alpha[r] + sum;
            mrow[r] = mn;
        }
        #pragma unroll
        for (int j = 0; j < 4; ++j)
            #pragma unroll
            for (int r = 0; r < 8; ++r) acc[j][r] *= alpha[r];
        #pragma unroll
        for (int r = 0; r < 8; ++r) {
            int row = r + 8 * half;
            pw[row * 32 + ln]      = f2bf(s0[r]);
            pw[row * 32 + 16 + ln] = f2bf(s1[r]);
        }
        __asm__ volatile("s_wait_dscnt 0" ::: "memory");
        bf16x16 ap;
        ap.h[0] = *(const v8bf*)(pw + ln * 32 + half * 8);
        ap.h[1] = *(const v8bf*)(pw + ln * 32 + 16 + half * 8);
        bf16x16 bv[4];
        #pragma unroll
        for (int j = 0; j < 4; ++j) {
            const __bf16* bp = vv + (j * 16 + ln) * VSTR + half * 8;
            bv[j].h[0] = *(const v8bf*)(bp);
            bv[j].h[1] = *(const v8bf*)(bp + 16);
        }
        #pragma unroll
        for (int j = 0; j < 4; ++j)
            acc[j] = __builtin_amdgcn_wmma_f32_16x16x32_bf16(
                false, ap.v, false, bv[j].v, (short)0, acc[j], false, false);
    }
    #pragma unroll
    for (int r = 0; r < 8; ++r) lrow[r] = 1.f / lrow[r];
    #pragma unroll
    for (int j = 0; j < 4; ++j)
        #pragma unroll
        for (int r = 0; r < 8; ++r) {
            size_t off = ((size_t)(b * SEQ + q0 + r + 8 * half)) * DIM + h * DH + j * 16 + ln;
            ctx[off] = f2bf(acc[j][r] * lrow[r]);
        }
}

// ---------------------------------------------------------------------------
// Orchestration
// ---------------------------------------------------------------------------
extern "C" void kernel_launch(void* const* d_in, const int* in_sizes, int n_in,
                              void* d_out, int out_size, void* d_ws, size_t ws_size,
                              hipStream_t stream) {
    (void)in_sizes; (void)n_in; (void)out_size; (void)ws_size;
    const float* x     = (const float*)d_in[0];
    const int*   amask = (const int*)d_in[1];
    const float* Wq = (const float*)d_in[2];  const float* bq = (const float*)d_in[3];
    const float* Wk = (const float*)d_in[4];  const float* bk = (const float*)d_in[5];
    const float* Wv = (const float*)d_in[6];  const float* bv = (const float*)d_in[7];
    const float* Wo = (const float*)d_in[8];  const float* bo = (const float*)d_in[9];
    const float* ln1_g = (const float*)d_in[10]; const float* ln1_b = (const float*)d_in[11];
    const float* W1 = (const float*)d_in[12]; const float* b1 = (const float*)d_in[13];
    const float* W2 = (const float*)d_in[14]; const float* b2 = (const float*)d_in[15];
    const float* ln2_g = (const float*)d_in[16]; const float* ln2_b = (const float*)d_in[17];

    const int M = BSZ * SEQ;
    char* ws = (char*)d_ws;
    size_t off = 0;
    auto carve = [&](size_t bytes) { char* p = ws + off; off = (off + bytes + 255) & ~(size_t)255; return p; };

    __bf16* x_bf   = (__bf16*)carve((size_t)M * DIM * 2);
    __bf16* wq_t   = (__bf16*)carve((size_t)DIM * DIM * 2);
    __bf16* wk_t   = (__bf16*)carve((size_t)DIM * DIM * 2);
    __bf16* wv_t   = (__bf16*)carve((size_t)DIM * DIM * 2);
    __bf16* wo_t   = (__bf16*)carve((size_t)DIM * DIM * 2);
    __bf16* w1_t   = (__bf16*)carve((size_t)FF * DIM * 2);
    __bf16* w2_t   = (__bf16*)carve((size_t)DIM * FF * 2);
    __bf16* q_bf   = (__bf16*)carve((size_t)M * DIM * 2);
    __bf16* k_bf   = (__bf16*)carve((size_t)M * DIM * 2);
    __bf16* v_bf   = (__bf16*)carve((size_t)M * DIM * 2);
    __bf16* vt_bf  = (__bf16*)carve((size_t)M * DIM * 2);
    __bf16* ctx_bf = (__bf16*)carve((size_t)M * DIM * 2);
    float*  sa_f   = (float*)carve((size_t)M * DIM * 4);
    float*  ln1_f  = (float*)carve((size_t)M * DIM * 4);
    __bf16* ln1_bf = (__bf16*)carve((size_t)M * DIM * 2);
    __bf16* h_bf   = (__bf16*)carve((size_t)M * FF * 2);
    float*  ffn_f  = (float*)carve((size_t)M * DIM * 4);

    cast_f32_bf16<<<(M * DIM) / 256, 256, 0, stream>>>(x, x_bf, M * DIM);
    transpose_cast<<<(DIM * DIM) / 256, 256, 0, stream>>>(Wq, wq_t, DIM, DIM);
    transpose_cast<<<(DIM * DIM) / 256, 256, 0, stream>>>(Wk, wk_t, DIM, DIM);
    transpose_cast<<<(DIM * DIM) / 256, 256, 0, stream>>>(Wv, wv_t, DIM, DIM);
    transpose_cast<<<(DIM * DIM) / 256, 256, 0, stream>>>(Wo, wo_t, DIM, DIM);
    transpose_cast<<<(DIM * FF) / 256, 256, 0, stream>>>(W1, w1_t, DIM, FF);
    transpose_cast<<<(FF * DIM) / 256, 256, 0, stream>>>(W2, w2_t, FF, DIM);

    dim3 gProj(DIM / 64, M / 128);
    gemm_bf16<<<gProj, 256, 0, stream>>>(x_bf, wq_t, bq, nullptr, q_bf, M, DIM, DIM, 0.125f, 0);
    gemm_bf16<<<gProj, 256, 0, stream>>>(x_bf, wk_t, bk, nullptr, k_bf, M, DIM, DIM, 1.0f, 0);
    gemm_bf16<<<gProj, 256, 0, stream>>>(x_bf, wv_t, bv, nullptr, v_bf, M, DIM, DIM, 1.0f, 0);
    transpose_v<<<(M * DIM) / 256, 256, 0, stream>>>(v_bf, vt_bf);

    attn_kernel<<<(BSZ * NH * (SEQ / 16)) / ATT_WARPS, ATT_WARPS * 32, 0, stream>>>(
        q_bf, k_bf, vt_bf, amask, ctx_bf);

    gemm_bf16<<<gProj, 256, 0, stream>>>(ctx_bf, wo_t, bo, sa_f, nullptr, M, DIM, DIM, 1.0f, 0);
    add_ln_kernel<<<M, 256, 0, stream>>>(sa_f, x, ln1_g, ln1_b, ln1_f, ln1_bf);

    dim3 gFF1(FF / 64, M / 128);
    gemm_bf16<<<gFF1, 256, 0, stream>>>(ln1_bf, w1_t, b1, nullptr, h_bf, M, FF, DIM, 1.0f, 1);
    gemm_bf16<<<gProj, 256, 0, stream>>>(h_bf, w2_t, b2, ffn_f, nullptr, M, DIM, FF, 1.0f, 0);
    add_ln_kernel<<<M, 256, 0, stream>>>(ffn_f, ln1_f, ln2_g, ln2_b, (float*)d_out, nullptr);
}
